// Actor_Critic_6116033429958
// MI455X (gfx1250) — matrix operand order repaired and checked by reading, not compiled, not run on hardware
//
#include <hip/hip_runtime.h>

typedef __attribute__((ext_vector_type(16))) _Float16 v16h;
typedef __attribute__((ext_vector_type(8)))  _Float16 v8h;
typedef __attribute__((ext_vector_type(8)))  float    v8f;

#define BATCH 32768
#define WAVES 4
#define SX 776   // X row stride (halfs): 768 used + 8 pad, 16B aligned rows
#define SY 520   // Y row stride (halfs): 512 used + 8 pad

// ---------- fast activations ----------
static __device__ __forceinline__ float fsig(float x) {
    return 1.0f / (1.0f + __expf(-x));
}
static __device__ __forceinline__ float ftanh(float x) {
    float t = __expf(-2.0f * fabsf(x));
    float r = (1.0f - t) / (1.0f + t);
    return copysignf(r, x);
}

// ---------- WMMA fragment helpers (CDNA5 16x16x32 f16) ----------
// A (16xK row-major in LDS): lane holds row (lane&15); K = half*16 + {0..15} contiguous
static __device__ __forceinline__ v16h ldA(const _Float16* p) {
    v8h lo = *(const v8h*)p;
    v8h hi = *(const v8h*)(p + 8);
    return __builtin_shufflevector(lo, hi, 0,1,2,3,4,5,6,7,8,9,10,11,12,13,14,15);
}
// B from [N][K] row-major f16 weights: lane holds col (lane&15); K = half*16 + {0..15} contiguous
static __device__ __forceinline__ v16h ldB(const _Float16* p) {
    v8h lo = *(const v8h*)p;
    v8h hi = *(const v8h*)(p + 8);
    return __builtin_shufflevector(lo, hi, 0,1,2,3,4,5,6,7,8,9,10,11,12,13,14,15);
}
static __device__ __forceinline__ v8f wm(v16h a, v16h b, v8f c) {
    return __builtin_amdgcn_wmma_f32_16x16x32_f16(false, a, false, b, (short)0, c, false, false);
}

// Accumulate one 16x16 output tile: A in LDS (row stride sA), W f16 global [N][sW], iterate K
static __device__ __forceinline__ v8f tile_acc(const _Float16* A, int sA,
                                               const _Float16* W, int sW, int K,
                                               int n0, int lm, int half, v8f acc) {
    const _Float16* arow = A + lm * sA + half * 16;
    const _Float16* wrow = W + (n0 + lm) * sW + half * 16;
    for (int kk = 0; kk < K; kk += 32)
        acc = wm(ldA(arow + kk), ldB(wrow + kk), acc);
    return acc;
}

// Store C tile (M = r + half*8, N = lane&15) to LDS as f16 with bias (+optional relu)
static __device__ __forceinline__ void store_lds(_Float16* D, int sD, int col0, v8f acc,
                                                 float bias, int lm, int half, int relu) {
#pragma unroll
    for (int r = 0; r < 8; r++) {
        float v = acc[r] + bias;
        if (relu) v = fmaxf(v, 0.0f);
        D[(r + half * 8) * sD + col0 + lm] = (_Float16)v;
    }
}

// ---------- pre-pass 1: weight f32 -> f16 ----------
__global__ void cvt_f16(const float* __restrict__ src, _Float16* __restrict__ dst, int n) {
    int i = blockIdx.x * 256 + threadIdx.x;
    if (i < n) dst[i] = (_Float16)src[i];
}

// ---------- pre-pass 2: gated_att has only 4 distinct rows (idx in 0..3) ----------
// tbl[idx][c] = sigmoid(emb[idx] . w_ta[c] + b_ta[c]),   4 x 256 floats
__global__ void gate_table_k(const float* __restrict__ emb, const float* __restrict__ w_ta,
                             const float* __restrict__ b_ta, float* __restrict__ tbl) {
    int c = threadIdx.x;  // 256 threads
    const float* wr = w_ta + c * 25;
    float b = b_ta[c];
#pragma unroll
    for (int idx = 0; idx < 4; idx++) {
        const float* er = emb + idx * 25;
        float acc = b;
        for (int k = 0; k < 25; k++) acc = fmaf(er[k], wr[k], acc);
        tbl[idx * 256 + c] = fsig(acc);
    }
}

// f16 workspace layout (element offsets)
#define O_WIH 0u
#define O_WHH 524288u
#define O_WBA 786432u
#define O_WQ  1048576u
#define O_WK  1114112u
#define O_WAT 1179648u
#define O_WP1 1310720u
#define O_WP2 1343488u
#define O_WV1 1351680u
#define O_WV2 1368064u
#define O_TBL_BYTES 2740224u   // = (1368064 + 2048) halfs * 2; 4x256 f32 table after weights

// ---------- fused actor-critic forward ----------
__global__ void __launch_bounds__(WAVES * 32)
actor_critic_fused(const float* __restrict__ img, const int* __restrict__ instr,
                   const float* __restrict__ hx_in, const float* __restrict__ cx_in,
                   const float* __restrict__ query, const float* __restrict__ att_tbl,
                   const _Float16* __restrict__ W16,
                   const float* __restrict__ b_ih, const float* __restrict__ b_hh,
                   const float* __restrict__ b_q, const float* __restrict__ b_k,
                   const float* __restrict__ b_ba, const float* __restrict__ b_at,
                   const float* __restrict__ b_p1, const float* __restrict__ b_p2,
                   const float* __restrict__ w_p, const float* __restrict__ b_p,
                   const float* __restrict__ b_v1, const float* __restrict__ b_v2,
                   const float* __restrict__ w_v, const float* __restrict__ b_v,
                   float* __restrict__ out) {
    extern __shared__ _Float16 smem[];
    const int tid = threadIdx.x;
    const int lane = tid & 31, wave = tid >> 5;
    const int lm = lane & 15, half = lane >> 4;

    _Float16* X = smem + wave * (16 * SX + 16 * SY);  // [16][SX]
    _Float16* Y = X + 16 * SX;                        // [16][SY]
    const int R = (blockIdx.x * WAVES + wave) * 16;   // global row base

    const _Float16* WIH = W16 + O_WIH;   // [1024][512]
    const _Float16* WHH = W16 + O_WHH;   // [1024][256]
    const _Float16* WBA = W16 + O_WBA;   // [512][512]
    const _Float16* WQ  = W16 + O_WQ;    // [256][256]
    const _Float16* WK  = W16 + O_WK;    // [256][256]
    const _Float16* WAT = W16 + O_WAT;   // [256][512]
    const _Float16* WP1 = W16 + O_WP1;   // [128][256]
    const _Float16* WP2 = W16 + O_WP2;   // [64][128]
    const _Float16* WV1 = W16 + O_WV1;   // [64][256]
    const _Float16* WV2 = W16 + O_WV2;   // [32][64]

    float* out_val = out;                                  // [B]
    float* out_pol = out + BATCH;                          // [B][3]
    float* out_hx  = out + (size_t)BATCH * 4;              // [B][256]
    float* out_cx  = out + (size_t)BATCH * 4 + (size_t)BATCH * 256;

    // ---- P0: gated_att via 4-row table, fusion; stage lstm_input in X, _hx(f16) in Y[:,0:256] ----
    for (int r = 0; r < 16; r++) {
        const int gr = R + r;
        const float* ar = att_tbl + instr[gr] * 256;
#pragma unroll
        for (int j = 0; j < 8; j++) {
            int c = lane + 32 * j;
            float att = ar[c];
            float fus = img[gr * 256 + c] * att;
            X[r * SX + c]       = (_Float16)fus;   // gated_fusion
            X[r * SX + 256 + c] = (_Float16)att;   // gated_att
            Y[r * SY + c]       = (_Float16)hx_in[gr * 256 + c];
        }
    }

    // ---- P1: LSTM gates (i,f,g,o share A fragments) -> hx/cx out + hx f16 -> Y[:,256:512] ----
    for (int n0 = 0; n0 < 256; n0 += 16) {
        v8f ai = {}, af = {}, ag = {}, ao = {};
        {   // K = 512 over lstm_input with w_ih
            const _Float16* arow = X + lm * SX + half * 16;
            const _Float16* w0 = WIH + (n0 +   0 + lm) * 512 + half * 16;
            const _Float16* w1 = WIH + (n0 + 256 + lm) * 512 + half * 16;
            const _Float16* w2 = WIH + (n0 + 512 + lm) * 512 + half * 16;
            const _Float16* w3 = WIH + (n0 + 768 + lm) * 512 + half * 16;
            for (int kk = 0; kk < 512; kk += 32) {
                v16h a = ldA(arow + kk);
                ai = wm(a, ldB(w0 + kk), ai);
                af = wm(a, ldB(w1 + kk), af);
                ag = wm(a, ldB(w2 + kk), ag);
                ao = wm(a, ldB(w3 + kk), ao);
            }
        }
        {   // K = 256 over _hx with w_hh
            const _Float16* arow = Y + lm * SY + half * 16;
            const _Float16* w0 = WHH + (n0 +   0 + lm) * 256 + half * 16;
            const _Float16* w1 = WHH + (n0 + 256 + lm) * 256 + half * 16;
            const _Float16* w2 = WHH + (n0 + 512 + lm) * 256 + half * 16;
            const _Float16* w3 = WHH + (n0 + 768 + lm) * 256 + half * 16;
            for (int kk = 0; kk < 256; kk += 32) {
                v16h a = ldA(arow + kk);
                ai = wm(a, ldB(w0 + kk), ai);
                af = wm(a, ldB(w1 + kk), af);
                ag = wm(a, ldB(w2 + kk), ag);
                ao = wm(a, ldB(w3 + kk), ao);
            }
        }
        const int col = n0 + lm;
        float bi = b_ih[col +   0] + b_hh[col +   0];
        float bf = b_ih[col + 256] + b_hh[col + 256];
        float bg = b_ih[col + 512] + b_hh[col + 512];
        float bo = b_ih[col + 768] + b_hh[col + 768];
#pragma unroll
        for (int r = 0; r < 8; r++) {
            int row = r + half * 8;
            int gr = R + row;
            float iv = fsig(ai[r] + bi);
            float fv = fsig(af[r] + bf);
            float gv = ftanh(ag[r] + bg);
            float ov = fsig(ao[r] + bo);
            float cxv = fv * cx_in[gr * 256 + col] + iv * gv;
            float hxv = ov * ftanh(cxv);
            out_cx[gr * 256 + col] = cxv;
            out_hx[gr * 256 + col] = hxv;
            Y[row * SY + 256 + col] = (_Float16)hxv;
        }
    }

    // ---- P2: X[:,256:512] <- hx (X becomes mlp_input = fusion|hx) ----
#pragma unroll
    for (int i = 0; i < 16; i++) {
        int cid = i * 32 + lane;
        int row = cid >> 5;
        int cc = (cid & 31) * 8;
        *(v8h*)(X + row * SX + 256 + cc) = *(const v8h*)(Y + row * SY + 256 + cc);
    }

    // ---- P3: mlp_attn = relu(mlp_input @ w_ba.T + b_ba); key->Y[:,0:256], val->X[:,512:768] ----
    for (int n0 = 0; n0 < 512; n0 += 16) {
        v8f z = {};
        v8f acc = tile_acc(X, SX, WBA, 512, 512, n0, lm, half, z);
        float bias = b_ba[n0 + lm];
        if (n0 < 256) {
            store_lds(Y, SY, n0, acc, bias, lm, half, 1);
        } else {
#pragma unroll
            for (int r = 0; r < 8; r++) {
                float v = fmaxf(acc[r] + bias, 0.0f);
                X[(r + half * 8) * SX + 512 + (n0 - 256) + lm] = (_Float16)v;
            }
        }
    }

    // ---- P4: query f16 -> X[:,0:256]; w_q_out = relu(query@w_q.T+b_q) -> X[:,256:512] ----
#pragma unroll
    for (int i = 0; i < 16; i++) {
        int cid = i * 32 + lane;
        int row = cid >> 5;
        int cc = (cid & 31) * 8;
        const float* qp = query + (size_t)(R + row) * 256 + cc;
        v8h h;
#pragma unroll
        for (int j = 0; j < 8; j++) h[j] = (_Float16)qp[j];
        *(v8h*)(X + row * SX + cc) = h;
    }
    for (int n0 = 0; n0 < 256; n0 += 16) {
        v8f z = {};
        v8f acc = tile_acc(X, SX, WQ, 256, 256, n0, lm, half, z);
        store_lds(X, SX, 256 + n0, acc, b_q[n0 + lm], lm, half, 1);
    }

    // ---- P5: w_k gemm on key_; u_t=tanh(wq+wk); att_vec = u_t*val -> X[:,0:256] ----
    for (int n0 = 0; n0 < 256; n0 += 16) {
        v8f z = {};
        v8f acc = tile_acc(Y, SY, WK, 256, 256, n0, lm, half, z);
        float bk = b_k[n0 + lm];
        int col = n0 + lm;
#pragma unroll
        for (int r = 0; r < 8; r++) {
            int row = r + half * 8;
            float wk = fmaxf(acc[r] + bk, 0.0f);
            float wq = (float)X[row * SX + 256 + col];
            float u = ftanh(wq + wk);
            float val = (float)X[row * SX + 512 + col];
            X[row * SX + col] = (_Float16)(u * val);
        }
    }

    // ---- P6: attn_weight = relu([att_vec|hx] @ w_at.T + b_at) -> X[:,256:512] ----
    for (int n0 = 0; n0 < 256; n0 += 16) {
        v8f z = {};
        v8f acc = tile_acc(X, SX, WAT, 512, 256, n0, lm, half, z);             // K 0:256 (att_vec)
        acc = tile_acc(Y + 256, SY, WAT + 256, 512, 256, n0, lm, half, acc);   // K 256:512 (hx)
        store_lds(X, SX, 256 + n0, acc, b_at[n0 + lm], lm, half, 1);
    }

    // ---- P7/P8/P9: policy head ----
    for (int n0 = 0; n0 < 128; n0 += 16) {  // pol1 -> Y[:,0:128]
        v8f z = {};
        v8f acc = tile_acc(X + 256, SX, WP1, 256, 256, n0, lm, half, z);
        store_lds(Y, SY, n0, acc, b_p1[n0 + lm], lm, half, 1);
    }
    for (int n0 = 0; n0 < 64; n0 += 16) {   // pol2 -> Y[:,128:192]
        v8f z = {};
        v8f acc = tile_acc(Y, SY, WP2, 128, 128, n0, lm, half, z);
        store_lds(Y, SY, 128 + n0, acc, b_p2[n0 + lm], lm, half, 1);
    }
    if (lane < 16) {                        // pol (N=3) scalar
        int row = lane, gr = R + row;
        for (int n = 0; n < 3; n++) {
            float acc = b_p[n];
            const float* wr = w_p + n * 64;
            for (int k = 0; k < 64; k++) acc = fmaf((float)Y[row * SY + 128 + k], wr[k], acc);
            out_pol[gr * 3 + n] = acc;
        }
    }

    // ---- P10/P11/P12: value head ----
    for (int n0 = 0; n0 < 64; n0 += 16) {   // v1 -> Y[:,192:256]
        v8f z = {};
        v8f acc = tile_acc(X + 256, SX, WV1, 256, 256, n0, lm, half, z);
        store_lds(Y, SY, 192 + n0, acc, b_v1[n0 + lm], lm, half, 1);
    }
    for (int n0 = 0; n0 < 32; n0 += 16) {   // v2 -> Y[:,256:288]
        v8f z = {};
        v8f acc = tile_acc(Y + 192, SY, WV2, 64, 64, n0, lm, half, z);
        store_lds(Y, SY, 256 + n0, acc, b_v2[n0 + lm], lm, half, 1);
    }
    if (lane < 16) {                        // value (N=1) scalar
        int row = lane, gr = R + row;
        float acc = b_v[0];
        for (int k = 0; k < 32; k++) acc = fmaf((float)Y[row * SY + 256 + k], w_v[k], acc);
        out_val[gr] = acc;
    }
}

extern "C" void kernel_launch(void* const* d_in, const int* in_sizes, int n_in,
                              void* d_out, int out_size, void* d_ws, size_t ws_size,
                              hipStream_t stream) {
    (void)in_sizes; (void)n_in; (void)out_size; (void)ws_size;
    const float* img   = (const float*)d_in[0];
    const int*   instr = (const int*)d_in[1];
    const float* hx    = (const float*)d_in[2];
    const float* cx    = (const float*)d_in[3];
    const float* query = (const float*)d_in[4];
    const float* emb   = (const float*)d_in[5];
    const float* w_ta  = (const float*)d_in[6];
    const float* b_ta  = (const float*)d_in[7];
    const float* w_ih  = (const float*)d_in[8];
    const float* w_hh  = (const float*)d_in[9];
    const float* b_ih  = (const float*)d_in[10];
    const float* b_hh  = (const float*)d_in[11];
    const float* w_q   = (const float*)d_in[12];
    const float* b_q   = (const float*)d_in[13];
    const float* w_k   = (const float*)d_in[14];
    const float* b_k   = (const float*)d_in[15];
    const float* w_ba  = (const float*)d_in[16];
    const float* b_ba  = (const float*)d_in[17];
    const float* w_at  = (const float*)d_in[18];
    const float* b_at  = (const float*)d_in[19];
    const float* w_p1  = (const float*)d_in[20];
    const float* b_p1  = (const float*)d_in[21];
    const float* w_p2  = (const float*)d_in[22];
    const float* b_p2  = (const float*)d_in[23];
    const float* w_p   = (const float*)d_in[24];
    const float* b_p   = (const float*)d_in[25];
    const float* w_v1  = (const float*)d_in[26];
    const float* b_v1  = (const float*)d_in[27];
    const float* w_v2  = (const float*)d_in[28];
    const float* b_v2  = (const float*)d_in[29];
    const float* w_v   = (const float*)d_in[30];
    const float* b_v   = (const float*)d_in[31];

    _Float16* W16 = (_Float16*)d_ws;
    float* att_tbl = (float*)((char*)d_ws + O_TBL_BYTES);

    auto cvt = [&](const float* s, unsigned off, int n) {
        cvt_f16<<<(n + 255) / 256, 256, 0, stream>>>(s, W16 + off, n);
    };
    cvt(w_ih, O_WIH, 1024 * 512);
    cvt(w_hh, O_WHH, 1024 * 256);
    cvt(w_ba, O_WBA, 512 * 512);
    cvt(w_q,  O_WQ,  256 * 256);
    cvt(w_k,  O_WK,  256 * 256);
    cvt(w_at, O_WAT, 256 * 512);
    cvt(w_p1, O_WP1, 128 * 256);
    cvt(w_p2, O_WP2, 64 * 128);
    cvt(w_v1, O_WV1, 64 * 256);
    cvt(w_v2, O_WV2, 32 * 64);
    gate_table_k<<<1, 256, 0, stream>>>(emb, w_ta, b_ta, att_tbl);

    const size_t lds = (size_t)WAVES * (16 * SX + 16 * SY) * sizeof(_Float16); // 165,888 B
    hipFuncSetAttribute((const void*)actor_critic_fused,
                        hipFuncAttributeMaxDynamicSharedMemorySize, (int)lds);
    actor_critic_fused<<<BATCH / (WAVES * 16), WAVES * 32, lds, stream>>>(
        img, instr, hx, cx, query, att_tbl, W16,
        b_ih, b_hh, b_q, b_k, b_ba, b_at, b_p1, b_p2, w_p, b_p,
        b_v1, b_v2, w_v, b_v, (float*)d_out);
}
